// GlobalAttentionBlock_72636486910357
// MI455X (gfx1250) — matrix operand order repaired and checked by reading, not compiled, not run on hardware
//
#include <hip/hip_runtime.h>
#include <hip/hip_bf16.h>
#include <math.h>

// ---------------------------------------------------------------------------
// GlobalAttentionBlock on gfx1250 (MI455X): all GEMMs via v_wmma_f32_16x16x32_f16
// B=1, H=W=48 -> N=2304 tokens, C=768, 12 heads x 64 head_dim.
// GEMM v2: 128x64 block tile, 32x32 wave tile (2x2 accums), ping-pong LDS with
// register prefetch -> 4 WMMAs per 8 ds_load_b128 in steady state.
// ---------------------------------------------------------------------------

typedef __attribute__((ext_vector_type(16))) _Float16 v16h;
typedef __attribute__((ext_vector_type(8)))  _Float16 v8h;
typedef __attribute__((ext_vector_type(8)))  float    v8f;

#define TOKENS   2304
#define CH       768
#define NHEADS   12
#define HDIM     64
#define GRID     48

#define BM 128
#define BN 64
#define BK 64
#define LDSW 72   // 64 + 8 f16 pad; row pitch 144B (16B aligned)

__device__ __forceinline__ float gelu_erf(float x) {
    return 0.5f * x * (1.0f + erff(x * 0.70710678118654752440f));
}

__device__ __forceinline__ v16h cat16(v8h lo, v8h hi) {
    return __builtin_shufflevector(lo, hi, 0,1,2,3,4,5,6,7,8,9,10,11,12,13,14,15);
}

// ---------------------------------------------------------------------------
// Generic WMMA GEMM: out = epilogue( scale * (A @ B^T) + bias + resid )
//   A : M x K f16, row-major, stride lda   (M multiple of 128)
//   Bt: N x K f16, row-major, stride ldb   (N multiple of 64, K multiple of 64)
// 256 threads = 8 waves; wave w -> rows (w&3)*32, cols (w>>2)*32 of the tile.
// EXEC all-ones everywhere around WMMA (no divergence).
// ---------------------------------------------------------------------------
__global__ __launch_bounds__(256)
void wmma_gemm_kernel(const _Float16* __restrict__ A, int lda,
                      const _Float16* __restrict__ Bt, int ldb,
                      int M, int N, int K,
                      float* __restrict__ outF, int ldo,
                      _Float16* __restrict__ outH, int ldoh,
                      const float* __restrict__ bias,
                      const float* __restrict__ resid, int ldr,
                      float scale, int apply_gelu)
{
    __shared__ __align__(16) _Float16 As[2][BM * LDSW];
    __shared__ __align__(16) _Float16 Bs[2][BN * LDSW];

    const int tid  = threadIdx.x;
    const int wave = tid >> 5;
    const int lane = tid & 31;
    const int m0 = blockIdx.y * BM;
    const int n0 = blockIdx.x * BN;

    const int wm = (wave & 3) * 32;   // 0,32,64,96
    const int wn = (wave >> 2) * 32;  // 0,32

    v8f c00 = {}, c01 = {}, c10 = {}, c11 = {};

    // cooperative tile load mapping: 256 threads, 16-f16 (2x b128) chunks
    const int lr = tid >> 2;           // 0..63
    const int lc = (tid & 3) * 16;     // 0,16,32,48

    // fragment coordinates (ISA 7.12.2 layouts)
    const int fm  = lane & 15;
    const int akb = (lane >> 4) * 8;   // A K-chunk base
    const int bkb = (lane >> 4) * 8;   // B K base

    v8h ar[4], br[2];
    auto load_tile = [&](int k0) {
        const _Float16* ag0 = A  + (size_t)(m0 + lr) * lda + (k0 + lc);
        const _Float16* ag1 = A  + (size_t)(m0 + 64 + lr) * lda + (k0 + lc);
        const _Float16* bg  = Bt + (size_t)(n0 + lr) * ldb + (k0 + lc);
        ar[0] = *(const v8h*)(ag0);
        ar[1] = *(const v8h*)(ag0 + 8);
        ar[2] = *(const v8h*)(ag1);
        ar[3] = *(const v8h*)(ag1 + 8);
        br[0] = *(const v8h*)(bg);
        br[1] = *(const v8h*)(bg + 8);
    };
    auto store_tile = [&](int s) {
        *(v8h*)(&As[s][lr * LDSW + lc])          = ar[0];
        *(v8h*)(&As[s][lr * LDSW + lc + 8])      = ar[1];
        *(v8h*)(&As[s][(lr + 64) * LDSW + lc])   = ar[2];
        *(v8h*)(&As[s][(lr + 64) * LDSW + lc + 8]) = ar[3];
        *(v8h*)(&Bs[s][lr * LDSW + lc])          = br[0];
        *(v8h*)(&Bs[s][lr * LDSW + lc + 8])      = br[1];
    };

    load_tile(0);
    store_tile(0);
    __syncthreads();

    int s = 0;
    for (int k0 = 0; k0 < K; k0 += BK) {
        const bool more = (k0 + BK) < K;
        if (more) load_tile(k0 + BK);   // prefetch next stage while computing

#pragma unroll
        for (int kk = 0; kk < BK; kk += 32) {
            // A fragments (rows wm..wm+15 and wm+16..wm+31)
            v16h a0 = cat16(*(const v8h*)(&As[s][(wm + fm)      * LDSW + kk + akb]),
                            *(const v8h*)(&As[s][(wm + fm)      * LDSW + kk + akb + 16]));
            v16h a1 = cat16(*(const v8h*)(&As[s][(wm + 16 + fm) * LDSW + kk + akb]),
                            *(const v8h*)(&As[s][(wm + 16 + fm) * LDSW + kk + akb + 16]));
            // B fragments (cols wn..wn+15 and wn+16..wn+31)
            v16h b0 = cat16(*(const v8h*)(&Bs[s][(wn + fm)      * LDSW + kk + bkb]),
                            *(const v8h*)(&Bs[s][(wn + fm)      * LDSW + kk + bkb + 16]));
            v16h b1 = cat16(*(const v8h*)(&Bs[s][(wn + 16 + fm) * LDSW + kk + bkb]),
                            *(const v8h*)(&Bs[s][(wn + 16 + fm) * LDSW + kk + bkb + 16]));

            c00 = __builtin_amdgcn_wmma_f32_16x16x32_f16(false, a0, false, b0, (short)0, c00, false, false);
            c01 = __builtin_amdgcn_wmma_f32_16x16x32_f16(false, a0, false, b1, (short)0, c01, false, false);
            c10 = __builtin_amdgcn_wmma_f32_16x16x32_f16(false, a1, false, b0, (short)0, c10, false, false);
            c11 = __builtin_amdgcn_wmma_f32_16x16x32_f16(false, a1, false, b1, (short)0, c11, false, false);
        }

        if (more) {
            store_tile(s ^ 1);
            __syncthreads();
        }
        s ^= 1;
    }

    // epilogue: C/D layout -> VGPR r: lanes 0-15 M=r, lanes 16-31 M=8+r
    const int ecol = lane & 15;
    const int erow = (lane >> 4) * 8;
    v8f* acc[2][2] = { { &c00, &c01 }, { &c10, &c11 } };
#pragma unroll
    for (int mi = 0; mi < 2; ++mi) {
#pragma unroll
        for (int ni = 0; ni < 2; ++ni) {
            v8f c = *acc[mi][ni];
#pragma unroll
            for (int r = 0; r < 8; ++r) {
                const int row = m0 + wm + mi * 16 + erow + r;
                const int col = n0 + wn + ni * 16 + ecol;
                float v = c[r] * scale;
                if (bias)  v += bias[col];
                if (resid) v += resid[(size_t)row * ldr + col];
                if (apply_gelu) v = gelu_erf(v);
                if (outF) outF[(size_t)row * ldo + col] = v;
                if (outH) outH[(size_t)row * ldoh + col] = (_Float16)v;
            }
        }
    }
}

// ---------------------------------------------------------------------------
// LayerNorm over C=768, one block (256 threads) per token row; f16 output.
// ---------------------------------------------------------------------------
__global__ __launch_bounds__(256)
void ln_kernel(const float* __restrict__ x, const float* __restrict__ g,
               const float* __restrict__ b, _Float16* __restrict__ out)
{
    const int row = blockIdx.x;
    const int tid = threadIdx.x;
    const float* xr = x + (size_t)row * CH;
    __shared__ float red[256];

    float v[3];
    float s = 0.f;
#pragma unroll
    for (int i = 0; i < 3; ++i) { v[i] = xr[tid + i * 256]; s += v[i]; }
    red[tid] = s; __syncthreads();
    for (int off = 128; off > 0; off >>= 1) {
        if (tid < off) red[tid] += red[tid + off];
        __syncthreads();
    }
    const float mean = red[0] * (1.0f / CH);
    __syncthreads();

    float s2 = 0.f;
#pragma unroll
    for (int i = 0; i < 3; ++i) { float d = v[i] - mean; s2 += d * d; }
    red[tid] = s2; __syncthreads();
    for (int off = 128; off > 0; off >>= 1) {
        if (tid < off) red[tid] += red[tid + off];
        __syncthreads();
    }
    const float inv = rsqrtf(red[0] * (1.0f / CH) + 1e-6f);

#pragma unroll
    for (int i = 0; i < 3; ++i) {
        const int c = tid + i * 256;
        out[(size_t)row * CH + c] = (_Float16)((v[i] - mean) * inv * g[c] + b[c]);
    }
}

// ---------------------------------------------------------------------------
// Decomposed rel-pos terms for one head:
//   relh[n][k1] = sum_c q[n][c] * rel_pos_h[h - k1 + 47][c]   (n=(h,w))
//   relw[n][k2] = sum_c q[n][c] * rel_pos_w[w - k2 + 47][c]
// ---------------------------------------------------------------------------
__global__ __launch_bounds__(256)
void rel_kernel(const _Float16* __restrict__ q,   // qkv16 + head*64, stride 2304
                const float* __restrict__ rph, const float* __restrict__ rpw,
                float* __restrict__ relh, float* __restrict__ relw)
{
    const int idx = blockIdx.x * 256 + threadIdx.x;
    if (idx >= TOKENS * 96) return;
    const int n = idx / 96;
    const int j = idx % 96;
    const int h = n / GRID;
    const int w = n % GRID;
    const _Float16* qr = q + (size_t)n * TOKENS;
    if (j < 48) {
        const float* t = rph + (size_t)(h - j + GRID - 1) * HDIM;
        float s = 0.f;
#pragma unroll 8
        for (int c = 0; c < HDIM; ++c) s += (float)qr[c] * t[c];
        relh[(size_t)n * GRID + j] = s;
    } else {
        const int k = j - 48;
        const float* t = rpw + (size_t)(w - k + GRID - 1) * HDIM;
        float s = 0.f;
#pragma unroll 8
        for (int c = 0; c < HDIM; ++c) s += (float)qr[c] * t[c];
        relw[(size_t)n * GRID + k] = s;
    }
}

// ---------------------------------------------------------------------------
// Softmax over one score row (2304) with rel-pos bias folded in; f16 probs out.
// ---------------------------------------------------------------------------
__global__ __launch_bounds__(256)
void softmax_kernel(const float* __restrict__ S, const float* __restrict__ relh,
                    const float* __restrict__ relw, _Float16* __restrict__ P)
{
    const int n   = blockIdx.x;
    const int tid = threadIdx.x;
    __shared__ float sh[96];
    __shared__ float red[256];
    if (tid < 48)       sh[tid] = relh[(size_t)n * GRID + tid];
    else if (tid < 96)  sh[tid] = relw[(size_t)n * GRID + (tid - 48)];
    __syncthreads();

    const float* Sr = S + (size_t)n * TOKENS;
    float vals[9];
    float vmax = -3.0e38f;
#pragma unroll
    for (int i = 0; i < 9; ++i) {
        const int m = tid + i * 256;
        float v = Sr[m] + sh[m / GRID] + sh[48 + m % GRID];
        vals[i] = v;
        vmax = fmaxf(vmax, v);
    }
    red[tid] = vmax; __syncthreads();
    for (int off = 128; off > 0; off >>= 1) {
        if (tid < off) red[tid] = fmaxf(red[tid], red[tid + off]);
        __syncthreads();
    }
    vmax = red[0]; __syncthreads();

    float sum = 0.f;
#pragma unroll
    for (int i = 0; i < 9; ++i) { vals[i] = __expf(vals[i] - vmax); sum += vals[i]; }
    red[tid] = sum; __syncthreads();
    for (int off = 128; off > 0; off >>= 1) {
        if (tid < off) red[tid] += red[tid + off];
        __syncthreads();
    }
    const float rs = 1.0f / red[0];
#pragma unroll
    for (int i = 0; i < 9; ++i)
        P[(size_t)n * TOKENS + tid + i * 256] = (_Float16)(vals[i] * rs);
}

// ---------------------------------------------------------------------------
// Small conversion / transpose helpers
// ---------------------------------------------------------------------------
__global__ void cvt_t_kernel(const float* __restrict__ src, _Float16* __restrict__ dst,
                             int R, int Ncol)   // dst[n][k] = src[k][n]
{
    const int idx = blockIdx.x * 256 + threadIdx.x;
    if (idx >= R * Ncol) return;
    const int k = idx / Ncol;
    const int n = idx % Ncol;
    dst[(size_t)n * R + k] = (_Float16)src[idx];
}

__global__ void cvt_kernel(const float* __restrict__ src, _Float16* __restrict__ dst, int n)
{
    const int idx = blockIdx.x * 256 + threadIdx.x;
    if (idx < n) dst[idx] = (_Float16)src[idx];
}

// vt[h*64+d][n] = qkv16[n][1536 + h*64 + d]  (all heads' V, transposed)
__global__ void vt_kernel(const _Float16* __restrict__ qkv, _Float16* __restrict__ vt)
{
    const int idx = blockIdx.x * 256 + threadIdx.x;
    if (idx >= CH * TOKENS) return;
    const int r = idx / TOKENS;
    const int n = idx % TOKENS;
    vt[(size_t)r * TOKENS + n] = qkv[(size_t)n * TOKENS + 2 * CH + r];
}

// ---------------------------------------------------------------------------
extern "C" void kernel_launch(void* const* d_in, const int* in_sizes, int n_in,
                              void* d_out, int out_size, void* d_ws, size_t ws_size,
                              hipStream_t stream)
{
    const float* x     = (const float*)d_in[0];
    const float* ln1_g = (const float*)d_in[1];
    const float* ln1_b = (const float*)d_in[2];
    const float* qkv_w = (const float*)d_in[3];
    const float* qkv_b = (const float*)d_in[4];
    const float* rph   = (const float*)d_in[5];
    const float* rpw   = (const float*)d_in[6];
    const float* proj_w= (const float*)d_in[7];
    const float* proj_b= (const float*)d_in[8];
    const float* ln2_g = (const float*)d_in[9];
    const float* ln2_b = (const float*)d_in[10];
    const float* mlp_w1= (const float*)d_in[11];
    const float* mlp_b1= (const float*)d_in[12];
    const float* mlp_w2= (const float*)d_in[13];
    const float* mlp_b2= (const float*)d_in[14];

    char* p = (char*)d_ws;
    auto alloc = [&](size_t bytes) -> void* {
        void* r = (void*)p;
        p += (bytes + 255) & ~(size_t)255;
        return r;
    };
    const size_t NT = TOKENS;
    _Float16* qkv_wt  = (_Float16*)alloc((size_t)3 * CH * CH * 2);   // (3C) x C
    _Float16* proj_wt = (_Float16*)alloc((size_t)CH * CH * 2);
    _Float16* w1t     = (_Float16*)alloc((size_t)4 * CH * CH * 2);
    _Float16* w2t     = (_Float16*)alloc((size_t)CH * 4 * CH * 2);
    _Float16* xn16    = (_Float16*)alloc(NT * CH * 2);
    _Float16* qkv16   = (_Float16*)alloc(NT * 3 * CH * 2);
    _Float16* vt16    = (_Float16*)alloc((size_t)CH * NT * 2);
    float*    S       = (float*)   alloc(NT * NT * 4);
    _Float16* P16     = (_Float16*)alloc(NT * NT * 2);
    float*    relh    = (float*)   alloc(NT * GRID * 4);
    float*    relw    = (float*)   alloc(NT * GRID * 4);
    float*    attnF   = (float*)   alloc(NT * CH * 4);
    _Float16* attn16  = (_Float16*)alloc(NT * CH * 2);
    float*    x1      = (float*)   alloc(NT * CH * 4);
    _Float16* y16     = (_Float16*)alloc(NT * CH * 2);
    _Float16* h16     = (_Float16*)alloc(NT * 4 * CH * 2);

    // --- weight converts (pre-transposed to N x K for the GEMM B operand) ---
    cvt_t_kernel<<<(CH * 3 * CH + 255) / 256, 256, 0, stream>>>(qkv_w,  qkv_wt,  CH, 3 * CH);
    cvt_t_kernel<<<(CH * CH + 255) / 256,     256, 0, stream>>>(proj_w, proj_wt, CH, CH);
    cvt_t_kernel<<<(CH * 4 * CH + 255) / 256, 256, 0, stream>>>(mlp_w1, w1t,     CH, 4 * CH);
    cvt_t_kernel<<<(4 * CH * CH + 255) / 256, 256, 0, stream>>>(mlp_w2, w2t,     4 * CH, CH);

    // --- LN1 -> QKV GEMM (f16 out, bias fused) ---
    ln_kernel<<<TOKENS, 256, 0, stream>>>(x, ln1_g, ln1_b, xn16);
    {
        dim3 g(3 * CH / BN, TOKENS / BM);
        wmma_gemm_kernel<<<g, 256, 0, stream>>>(xn16, CH, qkv_wt, CH,
            TOKENS, 3 * CH, CH, nullptr, 0, qkv16, 3 * CH, qkv_b, nullptr, 0, 1.f, 0);
    }
    vt_kernel<<<(CH * TOKENS + 255) / 256, 256, 0, stream>>>(qkv16, vt16);

    // --- per-head attention ---
    for (int hd = 0; hd < NHEADS; ++hd) {
        const _Float16* qh = qkv16 + hd * HDIM;            // 2304 x 64, stride 2304
        const _Float16* kh = qkv16 + CH + hd * HDIM;       // 2304 x 64, stride 2304
        rel_kernel<<<(TOKENS * 96 + 255) / 256, 256, 0, stream>>>(qh, rph, rpw, relh, relw);
        {
            dim3 g(TOKENS / BN, TOKENS / BM);  // S = 0.125 * q @ k^T
            wmma_gemm_kernel<<<g, 256, 0, stream>>>(qh, 3 * CH, kh, 3 * CH,
                TOKENS, TOKENS, HDIM, S, TOKENS, nullptr, 0, nullptr, nullptr, 0, 0.125f, 0);
        }
        softmax_kernel<<<TOKENS, 256, 0, stream>>>(S, relh, relw, P16);
        {
            dim3 g(HDIM / BN, TOKENS / BM);    // O_head = P @ v  (v^T precomputed)
            wmma_gemm_kernel<<<g, 256, 0, stream>>>(P16, TOKENS,
                vt16 + (size_t)hd * HDIM * TOKENS, TOKENS,
                TOKENS, HDIM, TOKENS, attnF + hd * HDIM, CH,
                nullptr, 0, nullptr, nullptr, 0, 1.f, 0);
        }
    }

    // --- proj + residual -> x1 ---
    cvt_kernel<<<(TOKENS * CH + 255) / 256, 256, 0, stream>>>(attnF, attn16, TOKENS * CH);
    {
        dim3 g(CH / BN, TOKENS / BM);
        wmma_gemm_kernel<<<g, 256, 0, stream>>>(attn16, CH, proj_wt, CH,
            TOKENS, CH, CH, x1, CH, nullptr, 0, proj_b, x, CH, 1.f, 0);
    }

    // --- LN2 -> MLP1 (GELU) -> MLP2 (+ residual) -> d_out ---
    ln_kernel<<<TOKENS, 256, 0, stream>>>(x1, ln2_g, ln2_b, y16);
    {
        dim3 g(4 * CH / BN, TOKENS / BM);
        wmma_gemm_kernel<<<g, 256, 0, stream>>>(y16, CH, w1t, CH,
            TOKENS, 4 * CH, CH, nullptr, 0, h16, 4 * CH, mlp_b1, nullptr, 0, 1.f, 1);
    }
    {
        dim3 g(CH / BN, TOKENS / BM);
        wmma_gemm_kernel<<<g, 256, 0, stream>>>(h16, 4 * CH, w2t, 4 * CH,
            TOKENS, CH, 4 * CH, (float*)d_out, CH, nullptr, 0, mlp_b2, x1, CH, 1.f, 0);
    }
}
